// GPT_4286377361594
// MI455X (gfx1250) — matrix operand order repaired and checked by reading, not compiled, not run on hardware
//
#include <hip/hip_runtime.h>
#include <stdint.h>

// ---------------------------------------------------------------------------
// MI455X (gfx1250) GPT forward. All GEMMs via v_wmma_f32_16x16x32_bf16.
// Tiles stream global->LDS with CDNA5 async-to-LDS DMA (ASYNCcnt) and
// double-buffered LDS; fragments read via ds_load_b128 in native WMMA layout.
// bf16 stored as raw uint16_t (RNE rounding in software).
// ---------------------------------------------------------------------------

typedef __attribute__((ext_vector_type(16))) __bf16    v16bf;
typedef __attribute__((ext_vector_type(8)))  float     v8f;
typedef __attribute__((ext_vector_type(8)))  uint16_t  ushort8;

union FragAB { v16bf v; ushort8 u[2]; };

__device__ __forceinline__ uint16_t f2bf(float f) {
    union { float f; uint32_t u; } c; c.f = f;
    uint32_t u = c.u;
    u += 0x7FFFu + ((u >> 16) & 1u);          // round-to-nearest-even
    return (uint16_t)(u >> 16);
}
__device__ __forceinline__ float bf2f(uint16_t h) {
    union { uint32_t u; float f; } c; c.u = ((uint32_t)h) << 16;
    return c.f;
}

// ---------------------------------------------------------------------------
// GEMM: C[M,N] = A[M,K](bf16 rm) * Bt[N,K](bf16 rm) (+bias)(+resid)(relu)
// Block = 256 threads (8 waves, 4x2), tile 256x128, K-step 32.
// Each wave owns a 64x64 patch = 4x4 WMMA accumulators (16 wmma : 16 ds_load).
// Batched via gridDim.z: offset = (z/batchH)*outer + (z%batchH)*inner.
// ---------------------------------------------------------------------------
#define BM 256
#define BN 128
#define BK 32
#define LDSS 40   // padded row stride (ushorts): 80B = 20 banks -> conflict-free b128 frag loads

template<bool BIAS, bool RELU, bool RESID, bool OUTBF16>
__global__ __launch_bounds__(256) void gemm_bf16_wmma(
    const uint16_t* __restrict__ A,  int lda, long long aOuter, long long aInner,
    const uint16_t* __restrict__ Bt, int ldb, long long bOuter, long long bInner,
    void* __restrict__ Cout,         int ldc, long long cOuter, long long cInner,
    const float* __restrict__ bias,
    const float* __restrict__ Resid, int ldr, long long rOuter, long long rInner,
    int M, int N, int K, int batchH)
{
    __shared__ uint16_t As[2][BM * LDSS];
    __shared__ uint16_t Bs[2][BN * LDSS];

    const int tid  = threadIdx.x;
    const int lane = tid & 31;
    const int wave = tid >> 5;
    const int wm   = wave >> 1;      // 0..3  (64 rows each)
    const int wn   = wave & 1;       // 0..1  (64 cols each)
    const int half = lane >> 4;      // selects K-subgroup / M-subrows per ISA layout
    const int l16  = lane & 15;

    const long long zb = (long long)(blockIdx.z) / batchH;
    const long long zi = (long long)(blockIdx.z) % batchH;
    A  += zb * aOuter + zi * aInner;
    Bt += zb * bOuter + zi * bInner;
    uint16_t* Cb = (uint16_t*)Cout + zb * cOuter + zi * cInner;
    float*    Cf = (float*)Cout    + zb * cOuter + zi * cInner;
    const float* Rp = RESID ? (Resid + zb * rOuter + zi * rInner) : nullptr;

    const int m0 = blockIdx.y * BM;
    const int n0 = blockIdx.x * BN;

    v8f acc[4][4] = {};

    // ---- async staging geometry ----
    // A: thread owns one full 64B tile row (4x b128). B: half row (2x b128).
    const int arow = tid;                   // 0..255
    const int brow = tid >> 1;              // 0..127
    const int bcol = (tid & 1) * 16;        // 0 or 16 (ushorts)
    const bool bvalid = (n0 + brow) < N;

    const uint16_t* gA = A  + (long long)(m0 + arow) * lda;
    const uint16_t* gB = Bt + (long long)(n0 + brow) * ldb + bcol;

    uint32_t ldsA[2], ldsB[2];
    ldsA[0] = (uint32_t)(uintptr_t)&As[0][arow * LDSS];
    ldsA[1] = (uint32_t)(uintptr_t)&As[1][arow * LDSS];
    ldsB[0] = (uint32_t)(uintptr_t)&Bs[0][brow * LDSS + bcol];
    ldsB[1] = (uint32_t)(uintptr_t)&Bs[1][brow * LDSS + bcol];

    auto prefetch = [&](int kt) {
        const int buf = kt & 1;
        const uint16_t* a = gA + kt * BK;
        asm volatile(
            "global_load_async_to_lds_b128 %0, %1, off\n\t"
            "global_load_async_to_lds_b128 %0, %1, off offset:16\n\t"
            "global_load_async_to_lds_b128 %0, %1, off offset:32\n\t"
            "global_load_async_to_lds_b128 %0, %1, off offset:48"
            :: "v"(ldsA[buf]), "v"(a) : "memory");
        if (bvalid) {
            const uint16_t* b = gB + kt * BK;
            asm volatile(
                "global_load_async_to_lds_b128 %0, %1, off\n\t"
                "global_load_async_to_lds_b128 %0, %1, off offset:16"
                :: "v"(ldsB[buf]), "v"(b) : "memory");
        }
    };

    // zero LDS rows of Bt that fall outside N (e.g. attention-O gemm, N=64):
    // async lanes with EXEC off never write them, so clear once for both buffers.
    if (!bvalid) {
        ushort8 z = {};
        *(ushort8*)(&Bs[0][brow * LDSS + bcol])     = z;
        *(ushort8*)(&Bs[0][brow * LDSS + bcol + 8]) = z;
        *(ushort8*)(&Bs[1][brow * LDSS + bcol])     = z;
        *(ushort8*)(&Bs[1][brow * LDSS + bcol + 8]) = z;
    }

    const int kTiles = K / BK;
    prefetch(0);
    asm volatile("s_wait_asynccnt 0x0" ::: "memory");
    __syncthreads();

    for (int kt = 0; kt < kTiles; ++kt) {
        const int cur = kt & 1;
        if (kt + 1 < kTiles) prefetch(kt + 1);

        // ---- fragments straight from LDS in native WMMA VGPR layout ----
        FragAB af[4], bfr[4];
        const uint16_t* ab = &As[cur][(wm * 64 + l16) * LDSS];
#pragma unroll
        for (int i = 0; i < 4; ++i) {
            const uint16_t* r = ab + i * 16 * LDSS;
            af[i].u[0] = *(const ushort8*)(r + half * 8);        // K 0..7  / 8..15
            af[i].u[1] = *(const ushort8*)(r + 16 + half * 8);   // K 16..23/ 24..31
        }
        const uint16_t* bb = &Bs[cur][(wn * 64 + l16) * LDSS];
#pragma unroll
        for (int j = 0; j < 4; ++j) {
            const uint16_t* r = bb + j * 16 * LDSS;
            bfr[j].u[0] = *(const ushort8*)(r + half * 8);       // K 0..7  / 8..15
            bfr[j].u[1] = *(const ushort8*)(r + 16 + half * 8);  // K 16..23/ 24..31
        }
#pragma unroll
        for (int i = 0; i < 4; ++i)
#pragma unroll
            for (int j = 0; j < 4; ++j)
                acc[i][j] = __builtin_amdgcn_wmma_f32_16x16x32_bf16(
                    false, af[i].v, false, bfr[j].v,
                    (short)0, acc[i][j], false, false);

        if (kt + 1 < kTiles) {
            asm volatile("s_wait_asynccnt 0x0" ::: "memory");  // next buffer landed
            __syncthreads();                                    // all waves done with cur
        }
    }

    // ---- epilogue: C layout = lane holds col (lane&15), VGPR r holds row (half*8+r)
#pragma unroll
    for (int i = 0; i < 4; ++i) {
#pragma unroll
        for (int j = 0; j < 4; ++j) {
            const int col = n0 + wn * 64 + j * 16 + l16;
            if (col >= N) continue;
            float bv = 0.f;
            if (BIAS) bv = bias[col];
#pragma unroll
            for (int r = 0; r < 8; ++r) {
                const int row = m0 + wm * 64 + i * 16 + half * 8 + r;
                float vv = acc[i][j][r] + bv;
                if (RESID) vv += Rp[(long long)row * ldr + col];
                if (RELU)  vv = fmaxf(vv, 0.f);
                if (OUTBF16) Cb[(long long)row * ldc + col] = f2bf(vv);
                else         Cf[(long long)row * ldc + col] = vv;
            }
        }
    }
}

// ---------------------------------------------------------------------------
// Embedding: x[row,:] = tok_emb[idx[row],:] + pos_emb[row%T,:]
// ---------------------------------------------------------------------------
__global__ __launch_bounds__(256) void embed_kernel(
    const int* __restrict__ idx, const float* __restrict__ tok,
    const float* __restrict__ pos, float* __restrict__ x, int D_, int T_)
{
    const long long row = blockIdx.x;
    const int t   = (int)(row % T_);
    const int tk  = idx[row];
    const int c   = threadIdx.x * 4;
    const float4 a = *(const float4*)(tok + (long long)tk * D_ + c);
    const float4 p = *(const float4*)(pos + (long long)t  * D_ + c);
    float4 o; o.x = a.x + p.x; o.y = a.y + p.y; o.z = a.z + p.z; o.w = a.w + p.w;
    *(float4*)(x + row * D_ + c) = o;
}

// ---------------------------------------------------------------------------
// LayerNorm over D=1024, block per row, bf16 output.
// ---------------------------------------------------------------------------
__global__ __launch_bounds__(256) void layernorm_bf16(
    const float* __restrict__ x, const float* __restrict__ s,
    const float* __restrict__ b, uint16_t* __restrict__ out, int D_)
{
    __shared__ float red[256];
    const long long row = blockIdx.x;
    const int tid = threadIdx.x;
    const float4 f4 = *(const float4*)(x + row * D_ + tid * 4);
    float v[4] = { f4.x, f4.y, f4.z, f4.w };

    float ss = v[0] + v[1] + v[2] + v[3];
    red[tid] = ss; __syncthreads();
    for (int o = 128; o > 0; o >>= 1) { if (tid < o) red[tid] += red[tid + o]; __syncthreads(); }
    const float mu = red[0] / (float)D_;
    __syncthreads();

    float s2 = 0.f;
#pragma unroll
    for (int c = 0; c < 4; ++c) { float d = v[c] - mu; s2 += d * d; }
    red[tid] = s2; __syncthreads();
    for (int o = 128; o > 0; o >>= 1) { if (tid < o) red[tid] += red[tid + o]; __syncthreads(); }
    const float inv = rsqrtf(red[0] / (float)D_ + 1e-5f);

#pragma unroll
    for (int c = 0; c < 4; ++c) {
        const int col = tid * 4 + c;
        out[row * D_ + col] = f2bf((v[c] - mu) * inv * s[col] + b[col]);
    }
}

// ---------------------------------------------------------------------------
// Causal softmax on one score row (len T=1024), scale 1/sqrt(64), bf16 out.
// grid = (T, B*H)
// ---------------------------------------------------------------------------
__global__ __launch_bounds__(256) void softmax_causal(
    const float* __restrict__ S, uint16_t* __restrict__ P, int T_)
{
    __shared__ float red[256];
    const int t = blockIdx.x;
    const long long base = ((long long)blockIdx.y * T_ + t) * T_;
    const int tid = threadIdx.x;
    const float scale = 0.125f;   // 1/sqrt(HD=64)

    const float4 f4 = *(const float4*)(S + base + tid * 4);
    float v[4] = { f4.x, f4.y, f4.z, f4.w };
    float mx = -3.4e38f;
#pragma unroll
    for (int c = 0; c < 4; ++c) {
        const int col = tid * 4 + c;
        v[c] = (col <= t) ? v[c] * scale : -3.4e38f;
        mx = fmaxf(mx, v[c]);
    }
    red[tid] = mx; __syncthreads();
    for (int o = 128; o > 0; o >>= 1) { if (tid < o) red[tid] = fmaxf(red[tid], red[tid + o]); __syncthreads(); }
    mx = red[0]; __syncthreads();

    float p[4], sum = 0.f;
#pragma unroll
    for (int c = 0; c < 4; ++c) {
        const int col = tid * 4 + c;
        p[c] = (col <= t) ? __expf(v[c] - mx) : 0.f;
        sum += p[c];
    }
    red[tid] = sum; __syncthreads();
    for (int o = 128; o > 0; o >>= 1) { if (tid < o) red[tid] += red[tid + o]; __syncthreads(); }
    const float rinv = 1.f / red[0];

#pragma unroll
    for (int c = 0; c < 4; ++c)
        P[base + tid * 4 + c] = f2bf(p[c] * rinv);
}

// ---------------------------------------------------------------------------
// Transpose + convert-to-bf16: in[R,C] (fp32 or bf16-as-ushort) -> out[C,R] bf16.
// Batched like the GEMM. Block (32,8), LDS 32x33.
// ---------------------------------------------------------------------------
__device__ __forceinline__ float ld_as_float(const float* p)    { return *p; }
__device__ __forceinline__ float ld_as_float(const uint16_t* p) { return bf2f(*p); }

template<typename Tin>
__global__ __launch_bounds__(256) void transpose_to_bf16(
    const Tin* __restrict__ in, int ldin, long long inOuter, long long inInner,
    uint16_t* __restrict__ out, int ldout, long long outOuter,
    int R, int C, int batchH)
{
    __shared__ float tile[32][33];
    const long long zb = (long long)(blockIdx.z) / batchH;
    const long long zi = (long long)(blockIdx.z) % batchH;
    in  += zb * inOuter + zi * inInner;
    out += (long long)blockIdx.z * outOuter;

    const int r0 = blockIdx.y * 32, c0 = blockIdx.x * 32;
    for (int i = threadIdx.y; i < 32; i += 8) {
        const int r = r0 + i, c = c0 + threadIdx.x;
        float v = 0.f;
        if (r < R && c < C) v = ld_as_float(in + (long long)r * ldin + c);
        tile[i][threadIdx.x] = v;
    }
    __syncthreads();
    for (int i = threadIdx.y; i < 32; i += 8) {
        const int orow = c0 + i, ocol = r0 + threadIdx.x;   // out row = original col
        if (orow < C && ocol < R)
            out[(long long)orow * ldout + ocol] = f2bf(tile[threadIdx.x][i]);
    }
}

// ---------------------------------------------------------------------------
// Host orchestration
// ---------------------------------------------------------------------------
extern "C" void kernel_launch(void* const* d_in, const int* in_sizes, int n_in,
                              void* d_out, int out_size, void* d_ws, size_t ws_size,
                              hipStream_t stream)
{
    (void)in_sizes; (void)n_in; (void)out_size; (void)ws_size;
    const int Bb = 2, T = 1024, Dm = 1024, Hh = 16, NL = 4, Vv = 32000, D4 = 4096;
    const long long BT = (long long)Bb * T;          // 2048 rows

    const int*   idx   = (const int*)  d_in[0];
    const float* tok   = (const float*)d_in[1];
    const float* pos   = (const float*)d_in[2];
    const float* Wq    = (const float*)d_in[3];
    const float* Wk    = (const float*)d_in[4];
    const float* Wv    = (const float*)d_in[5];
    const float* Wo    = (const float*)d_in[6];
    const float* bo    = (const float*)d_in[7];
    const float* ln1s  = (const float*)d_in[8];
    const float* ln1b  = (const float*)d_in[9];
    const float* ln2s  = (const float*)d_in[10];
    const float* ln2b  = (const float*)d_in[11];
    const float* W1    = (const float*)d_in[12];
    const float* b1    = (const float*)d_in[13];
    const float* W2    = (const float*)d_in[14];
    const float* b2    = (const float*)d_in[15];
    const float* lnfs  = (const float*)d_in[16];
    const float* lnfb  = (const float*)d_in[17];
    const float* Wlm   = (const float*)d_in[18];
    const float* blm   = (const float*)d_in[19];

    // ---- workspace carve (≈310 MiB; assumes d_ws is sized for it) ----
    size_t off = 0;
    char* base = (char*)d_ws;
    auto alloc = [&](size_t bytes) -> void* {
        void* p = base + off;
        off += (bytes + 255) & ~(size_t)255;
        return p;
    };
    float*    x    = (float*)   alloc((size_t)BT * Dm * 4);          // residual stream
    uint16_t* h    = (uint16_t*)alloc((size_t)BT * Dm * 2);          // LN out (bf16)
    uint16_t* q    = (uint16_t*)alloc((size_t)BT * Dm * 2);
    uint16_t* kbuf = (uint16_t*)alloc((size_t)BT * Dm * 2);
    uint16_t* vbuf = (uint16_t*)alloc((size_t)BT * Dm * 2);
    uint16_t* obuf = (uint16_t*)alloc((size_t)BT * Dm * 2);
    uint16_t* mmid = (uint16_t*)alloc((size_t)BT * D4 * 2);          // MLP hidden
    float*    Sbuf = (float*)   alloc((size_t)Bb * Hh * T * T * 4);  // scores (128 MiB)
    uint16_t* Pbuf = (uint16_t*)alloc((size_t)Bb * Hh * T * T * 2);  // softmax (64 MiB)
    uint16_t* Vt   = (uint16_t*)alloc((size_t)Bb * Hh * 64 * T * 2); // V^T per (b,h)
    uint16_t* wT   = (uint16_t*)alloc((size_t)Dm * D4 * 2);          // reusable W^T (8 MiB)
    uint16_t* WlmT = (uint16_t*)alloc((size_t)Vv * Dm * 2);          // LM head W^T (64 MiB)

    const dim3 blk(256), tblk(32, 8);

    // x = tok_emb[idx] + pos_emb
    embed_kernel<<<dim3((unsigned)BT), blk, 0, stream>>>(idx, tok, pos, x, Dm, T);

    for (int l = 0; l < NL; ++l) {
        const long long wDD = (long long)l * Dm * Dm;

        // h = LN1(x)
        layernorm_bf16<<<dim3((unsigned)BT), blk, 0, stream>>>(x, ln1s + (long long)l * Dm, ln1b + (long long)l * Dm, h, Dm);

        // Q = h @ Wq   (Bt = Wq^T)
        transpose_to_bf16<float><<<dim3(Dm/32, Dm/32, 1), tblk, 0, stream>>>(Wq + wDD, Dm, 0, 0, wT, Dm, 0, Dm, Dm, 1);
        gemm_bf16_wmma<false,false,false,true><<<dim3(Dm/BN, BT/BM, 1), blk, 0, stream>>>(
            h, Dm, 0, 0,  wT, Dm, 0, 0,  q, Dm, 0, 0,
            nullptr, nullptr, 0, 0, 0,  (int)BT, Dm, Dm, 1);
        // K
        transpose_to_bf16<float><<<dim3(Dm/32, Dm/32, 1), tblk, 0, stream>>>(Wk + wDD, Dm, 0, 0, wT, Dm, 0, Dm, Dm, 1);
        gemm_bf16_wmma<false,false,false,true><<<dim3(Dm/BN, BT/BM, 1), blk, 0, stream>>>(
            h, Dm, 0, 0,  wT, Dm, 0, 0,  kbuf, Dm, 0, 0,
            nullptr, nullptr, 0, 0, 0,  (int)BT, Dm, Dm, 1);
        // V
        transpose_to_bf16<float><<<dim3(Dm/32, Dm/32, 1), tblk, 0, stream>>>(Wv + wDD, Dm, 0, 0, wT, Dm, 0, Dm, Dm, 1);
        gemm_bf16_wmma<false,false,false,true><<<dim3(Dm/BN, BT/BM, 1), blk, 0, stream>>>(
            h, Dm, 0, 0,  wT, Dm, 0, 0,  vbuf, Dm, 0, 0,
            nullptr, nullptr, 0, 0, 0,  (int)BT, Dm, Dm, 1);

        // S[b,h] = Q[b,:,h*64:+64] @ K[b,:,h*64:+64]^T     (batched: z = b*H+h)
        gemm_bf16_wmma<false,false,false,false><<<dim3(T/BN, T/BM, Bb*Hh), blk, 0, stream>>>(
            q,    Dm, (long long)T * Dm, 64,
            kbuf, Dm, (long long)T * Dm, 64,
            Sbuf, T,  (long long)Hh * T * T, (long long)T * T,
            nullptr, nullptr, 0, 0, 0,  T, T, 64, Hh);

        // P = causal softmax(S / 8)
        softmax_causal<<<dim3(T, Bb*Hh), blk, 0, stream>>>(Sbuf, Pbuf, T);

        // Vt[b,h] = V[b,:,h*64:+64]^T  (bf16 -> bf16 transpose, batched)
        transpose_to_bf16<uint16_t><<<dim3(64/32, T/32, Bb*Hh), tblk, 0, stream>>>(
            vbuf, Dm, (long long)T * Dm, 64,  Vt, T, (long long)64 * T,  T, 64, Hh);

        // O[b,:,h*64:+64] = P[b,h] @ V[b,h]   (Bt = Vt, N=64 handled by guards)
        gemm_bf16_wmma<false,false,false,true><<<dim3(1, T/BM, Bb*Hh), blk, 0, stream>>>(
            Pbuf, T, (long long)Hh * T * T, (long long)T * T,
            Vt,   T, (long long)Hh * 64 * T, (long long)64 * T,
            obuf, Dm, (long long)T * Dm, 64,
            nullptr, nullptr, 0, 0, 0,  T, 64, T, Hh);

        // x = x + O @ Wo + bo
        transpose_to_bf16<float><<<dim3(Dm/32, Dm/32, 1), tblk, 0, stream>>>(Wo + wDD, Dm, 0, 0, wT, Dm, 0, Dm, Dm, 1);
        gemm_bf16_wmma<true,false,true,false><<<dim3(Dm/BN, BT/BM, 1), blk, 0, stream>>>(
            obuf, Dm, 0, 0,  wT, Dm, 0, 0,  x, Dm, 0, 0,
            bo + (long long)l * Dm,  x, Dm, 0, 0,  (int)BT, Dm, Dm, 1);

        // h = LN2(x)
        layernorm_bf16<<<dim3((unsigned)BT), blk, 0, stream>>>(x, ln2s + (long long)l * Dm, ln2b + (long long)l * Dm, h, Dm);

        // mmid = relu(h @ W1 + b1)
        transpose_to_bf16<float><<<dim3(D4/32, Dm/32, 1), tblk, 0, stream>>>(
            W1 + (long long)l * Dm * D4, D4, 0, 0, wT, Dm, 0, Dm, D4, 1);
        gemm_bf16_wmma<true,true,false,true><<<dim3(D4/BN, BT/BM, 1), blk, 0, stream>>>(
            h, Dm, 0, 0,  wT, Dm, 0, 0,  mmid, D4, 0, 0,
            b1 + (long long)l * D4, nullptr, 0, 0, 0,  (int)BT, D4, Dm, 1);

        // x = x + mmid @ W2 + b2
        transpose_to_bf16<float><<<dim3(Dm/32, D4/32, 1), tblk, 0, stream>>>(
            W2 + (long long)l * D4 * Dm, Dm, 0, 0, wT, D4, 0, D4, Dm, 1);
        gemm_bf16_wmma<true,false,true,false><<<dim3(Dm/BN, BT/BM, 1), blk, 0, stream>>>(
            mmid, D4, 0, 0,  wT, D4, 0, 0,  x, Dm, 0, 0,
            b2 + (long long)l * Dm,  x, Dm, 0, 0,  (int)BT, Dm, D4, 1);
    }

    // h = LN_f(x)
    layernorm_bf16<<<dim3((unsigned)BT), blk, 0, stream>>>(x, lnfs, lnfb, h, Dm);

    // logits = h @ Wlm + blm  -> d_out (fp32)
    transpose_to_bf16<float><<<dim3(Vv/32, Dm/32, 1), tblk, 0, stream>>>(Wlm, Vv, 0, 0, WlmT, Dm, 0, Dm, Vv, 1);
    gemm_bf16_wmma<true,false,false,false><<<dim3(Vv/BN, BT/BM, 1), blk, 0, stream>>>(
        h, Dm, 0, 0,  WlmT, Dm, 0, 0,  (float*)d_out, Vv, 0, 0,
        blm, nullptr, 0, 0, 0,  (int)BT, Vv, Dm, 1);
}
